// BasicAttn_32014686224792
// MI455X (gfx1250) — matrix operand order repaired and checked by reading, not compiled, not run on hardware
//
#include <hip/hip_runtime.h>

// CDNA5 / gfx1250 flash attention: [4,16,2048,64] fp32 in/out.
// f16 WMMA (V_WMMA_F32_16X16X32_F16), fp32 online softmax.
// 64-key blocks: 16 WMMAs per iteration vs per-iteration softmax overhead.

typedef __attribute__((ext_vector_type(16))) _Float16 v16h;
typedef __attribute__((ext_vector_type(8)))  float    v8f;

#define SEQ   2048
#define DK    64
#define KBLK  64
#define WAVES 4

// LDS row strides in halves: 68 halves = 136B = 34 dwords, gcd(34,64)=2 -> ~2-way banks
#define KT_STRIDE 68   // kT[64][68]   K transposed [d][key]
#define V_STRIDE  68   // vS[64][68]   V [key][d]
#define P_STRIDE  68   // pS[w][16][68] P staging [row][key]

union H16 { v16h h; unsigned int u[8]; };

__global__ __launch_bounds__(128)
void attn_flash_f16wmma(const float* __restrict__ Q,
                        const float* __restrict__ K,
                        const float* __restrict__ V,
                        float* __restrict__ O) {
    __shared__ _Float16 kT[DK][KT_STRIDE];
    __shared__ _Float16 vS[KBLK][V_STRIDE];
    __shared__ _Float16 pS[WAVES][16][P_STRIDE];

    const int tid  = threadIdx.x;
    const int lane = tid & 31;
    const int wave = tid >> 5;
    const int bh   = blockIdx.x >> 5;   // 64 (b,h)
    const int qblk = blockIdx.x & 31;   // 32 blocks of 64 queries
    const int qbase = qblk * 64 + wave * 16;
    const size_t base = (size_t)bh * SEQ * DK;

    // A-layout (16-bit 16x32): lane L -> row M = L&15;
    // lanes<16 hold K {0..7,16..23}, lanes>=16 hold K {8..15,24..31}.
    const int arow = lane & 15;
    const int koff = (lane >> 4) * 8;

    // ---- Q tile (16x64) -> two f16 A-register tiles, 1/sqrt(d) folded in ----
    H16 aq[2];
    {
        const float* qrow = Q + base + (size_t)(qbase + arow) * DK;
        #pragma unroll
        for (int c = 0; c < 2; ++c) {
            #pragma unroll
            for (int e = 0; e < 16; ++e) {
                int k = (e < 8) ? (koff + e) : (16 + koff + (e - 8));
                aq[c].h[e] = (_Float16)(qrow[c * 32 + k] * 0.125f);
            }
        }
    }

    v8f acc[4];
    #pragma unroll
    for (int dt = 0; dt < 4; ++dt) acc[dt] = (v8f){};
    float m[8], l[8];   // running stats for row r + 8*(lane>=16)
    #pragma unroll
    for (int r = 0; r < 8; ++r) { m[r] = -1e30f; l[r] = 0.0f; }

    for (int kb = 0; kb < SEQ / KBLK; ++kb) {
        __syncthreads();  // prior iteration done reading LDS
        {   // cooperative K/V block load (f32 -> f16; K stored transposed)
            const float* kp = K + base + (size_t)kb * KBLK * DK;
            const float* vp = V + base + (size_t)kb * KBLK * DK;
            for (int i = tid; i < KBLK * DK; i += 128) {
                int key = i >> 6, d = i & 63;
                kT[d][key] = (_Float16)kp[i];
                vS[key][d] = (_Float16)vp[i];
            }
            if (kb + 1 < SEQ / KBLK) {  // global_prefetch_b8 for next block
                __builtin_prefetch(kp + KBLK * DK + tid * 32, 0, 1);
                __builtin_prefetch(vp + KBLK * DK + tid * 32, 0, 1);
            }
        }
        __syncthreads();

        // ---- scores: S(16q x 64k), 4 C-tiles, 8 WMMAs ----
        // B-layout (16-bit 32x16): lane&15 = key col, elements = d in A's table order.
        v8f s[4];
        #pragma unroll
        for (int j = 0; j < 4; ++j) {
            H16 b0, b1;
            #pragma unroll
            for (int e = 0; e < 16; ++e) {
                int k = (e < 8) ? (koff + e) : (16 + koff + (e - 8));
                b0.h[e] = kT[k][16 * j + arow]; b1.h[e] = kT[32 + k][16 * j + arow];
            }
            v8f z = (v8f){};
            v8f t = __builtin_amdgcn_wmma_f32_16x16x32_f16(false, aq[0].h, false, b0.h,
                                                           (short)0, z, false, false);
            t     = __builtin_amdgcn_wmma_f32_16x16x32_f16(false, aq[1].h, false, b1.h,
                                                           (short)0, t, false, false);
            s[j] = t;
        }

        // ---- online softmax (row values live across 16-lane groups) ----
        float alpha[8];
        #pragma unroll
        for (int r = 0; r < 8; ++r) {
            float a0 = s[0][r], a1 = s[1][r], a2 = s[2][r], a3 = s[3][r];
            float t = fmaxf(fmaxf(a0, a1), fmaxf(a2, a3));
            t = fmaxf(t, __shfl_xor(t, 1, 32));
            t = fmaxf(t, __shfl_xor(t, 2, 32));
            t = fmaxf(t, __shfl_xor(t, 4, 32));
            t = fmaxf(t, __shfl_xor(t, 8, 32));
            float mn = fmaxf(m[r], t);
            alpha[r] = __expf(m[r] - mn);
            float p0 = __expf(a0 - mn), p1 = __expf(a1 - mn);
            float p2 = __expf(a2 - mn), p3 = __expf(a3 - mn);
            s[0][r] = p0; s[1][r] = p1; s[2][r] = p2; s[3][r] = p3;
            float rs = (p0 + p1) + (p2 + p3);
            rs += __shfl_xor(rs, 1, 32);
            rs += __shfl_xor(rs, 2, 32);
            rs += __shfl_xor(rs, 4, 32);
            rs += __shfl_xor(rs, 8, 32);
            l[r] = l[r] * alpha[r] + rs;
            m[r] = mn;
        }
        #pragma unroll
        for (int dt = 0; dt < 4; ++dt)
            #pragma unroll
            for (int r = 0; r < 8; ++r) acc[dt][r] = acc[dt][r] * alpha[r];

        // ---- stage P: C-layout f32 -> LDS row-major f16 [row][key] ----
        {
            int rb = (lane >> 4) * 8, nc = lane & 15;
            #pragma unroll
            for (int r = 0; r < 8; ++r)
                #pragma unroll
                for (int j = 0; j < 4; ++j)
                    pS[wave][rb + r][16 * j + nc] = (_Float16)s[j][r];
        }
        __syncthreads();

        // ---- O += P(16x64) x V(64x64): 2 key-chunks x 4 d-tiles = 8 WMMAs ----
        #pragma unroll
        for (int kc = 0; kc < 2; ++kc) {
            H16 ap;
            const unsigned int* pr = (const unsigned int*)&pS[wave][arow][0];
            #pragma unroll
            for (int v = 0; v < 4; ++v) ap.u[v]     = pr[kc * 16 + koff / 2 + v];
            #pragma unroll
            for (int v = 0; v < 4; ++v) ap.u[4 + v] = pr[kc * 16 + 8 + koff / 2 + v];
            #pragma unroll
            for (int dt = 0; dt < 4; ++dt) {
                H16 bv;
                #pragma unroll
                for (int e = 0; e < 16; ++e) {
                    int k = (e < 8) ? (koff + e) : (16 + koff + (e - 8));
                    bv.h[e] = vS[kc * 32 + k][dt * 16 + arow];
                }
                acc[dt] = __builtin_amdgcn_wmma_f32_16x16x32_f16(false, ap.h, false, bv.h,
                                                                 (short)0, acc[dt], false, false);
            }
        }
    }

    // ---- normalize and store (C-layout: lane = d column, vgpr = query row) ----
    float* orow = O + base;
    float inv[8];
    #pragma unroll
    for (int r = 0; r < 8; ++r) inv[r] = 1.0f / l[r];
    #pragma unroll
    for (int dt = 0; dt < 4; ++dt) {
        int d = dt * 16 + (lane & 15);
        #pragma unroll
        for (int r = 0; r < 8; ++r) {
            int rr = r + ((lane >> 4) * 8);
            orow[(size_t)(qbase + rr) * DK + d] = acc[dt][r] * inv[r];
        }
    }
}

extern "C" void kernel_launch(void* const* d_in, const int* in_sizes, int n_in,
                              void* d_out, int out_size, void* d_ws, size_t ws_size,
                              hipStream_t stream) {
    (void)in_sizes; (void)n_in; (void)out_size; (void)d_ws; (void)ws_size;
    const float* Q = (const float*)d_in[0];
    const float* K = (const float*)d_in[1];
    const float* V = (const float*)d_in[2];
    float* O = (float*)d_out;
    dim3 grid(64 * 32);   // 64 (b,h) * 32 query-blocks of 64 rows
    dim3 block(128);      // 4 wave32 waves
    hipLaunchKernelGGL(attn_flash_f16wmma, grid, block, 0, stream, Q, K, V, O);
}
